// ANet_70300024701046
// MI455X (gfx1250) — compile-verified, matrix operand order checked by reading, not run on hardware
//
#include <hip/hip_runtime.h>
#include <hip/hip_bf16.h>
#include <math.h>

typedef __attribute__((ext_vector_type(16))) _Float16 v16h;
typedef __attribute__((ext_vector_type(8)))  _Float16 v8h;
typedef __attribute__((ext_vector_type(8)))  float    v8f;
typedef __attribute__((ext_vector_type(4)))  unsigned int v4u;
typedef __attribute__((ext_vector_type(8)))  int      v8i;
typedef __attribute__((ext_vector_type(4)))  int      v4i;

#define LEAKY_SLOPE 0.01f
#define FLAG_RES   1
#define FLAG_LEAKY 2

#if defined(__has_builtin)
#if __has_builtin(__builtin_amdgcn_tensor_load_to_lds)
#define HAVE_TDM 1
#endif
#endif

#ifdef HAVE_TDM
// Stage a 2D weight tile (tile1 rows x tile0 f16 cols, row stride stride0 elems)
// from global into LDS at lds_off via the Tensor Data Mover, then drain TENSORcnt.
// D# bit packing per CDNA5 ISA ch.8 (group0: count/lds/global/type, group1: dims).
__device__ __forceinline__ void tdm_load_weights(unsigned lds_off, const _Float16* gptr,
                                                 int tile0, int tile1, int stride0)
{
    unsigned long long ga = (unsigned long long)(const void*)gptr;
    v4u g0;
    g0[0] = 1u;                                             // count=1 (valid user D#)
    g0[1] = lds_off;                                        // lds_addr (bytes)
    g0[2] = (unsigned)(ga & 0xFFFFFFFFu);                   // global_addr[31:0]
    g0[3] = (unsigned)((ga >> 32) & 0x1FFFFFFu) | (2u << 30); // global_addr[56:32], type=2
    v8i g1;
    g1[0] = (int)(1u << 16);                                // data_size=1 (2B), wg_mask=0
    g1[1] = (int)((unsigned)(tile0 & 0xFFFF) << 16);        // tensor_dim0[15:0]
    g1[2] = (int)((unsigned)(tile1 & 0xFFFF) << 16);        // tensor_dim1[15:0]
    g1[3] = (int)((unsigned)(tile0 & 0xFFFF) << 16);        // tile_dim0
    g1[4] = (int)(tile1 & 0xFFFF);                          // tile_dim1 (tile_dim2=0)
    g1[5] = stride0;                                        // tensor_dim0_stride[31:0]
    g1[6] = 0;                                              // stride hi / dim1_stride lo
    g1[7] = 0;
    v4i z4 = {0, 0, 0, 0};
#if __clang_major__ >= 23
    v8i z8 = {0, 0, 0, 0, 0, 0, 0, 0};
    __builtin_amdgcn_tensor_load_to_lds(g0, g1, z4, z4, z8, 0);
#else
    __builtin_amdgcn_tensor_load_to_lds(g0, g1, z4, z4, 0);
#endif
    __builtin_amdgcn_s_wait_tensorcnt(0);
}
#endif

// ---------------------------------------------------------------------------
// Implicit-GEMM conv on the matrix pipe: v_wmma_f32_16x16x32_f16.
// 256-thread blocks = 8 waves; each wave owns a 16(M) x O tile. The per-tap
// weight slab (O x CIN f16, <= 8 KB) is TDM-staged into LDS once per block
// and shared by all 8 waves (ds_load_b128 fragment reads): 8x less B traffic.
// A  : [M, CIN] f16 activations, channel-innermost
// Wt : [O, TAPS*CIN] f16 weights, K contiguous per output channel
// Y  : [M, O] f32
// MODE: 0 = 1x1, 1 = 3x3 zero pad, 2 = 3x3 reflect pad,
//       3 = stride-2 transposed conv (input is HD/2 x WD/2)
// ---------------------------------------------------------------------------
template <int MODE, int CIN, int O, int HD, int WD>
__global__ void __launch_bounds__(256) conv_wmma_t(
    const _Float16* __restrict__ A, const _Float16* __restrict__ Wt,
    const float* __restrict__ bias, const float* __restrict__ resid,
    float* __restrict__ Y, int flags)
{
    constexpr int NT   = O / 16;
    constexpr int TAPS = (MODE == 0) ? 1 : 9;
    constexpr int KTOT = TAPS * CIN;
    constexpr int P    = HD * WD;
    constexpr int HIN  = (MODE == 3) ? (HD / 2) : HD;
    constexpr int WIN  = (MODE == 3) ? (WD / 2) : WD;

    __shared__ _Float16 lw[O * CIN];      // one tap's weight slab

    const int tid  = threadIdx.x;
    const int wave = tid >> 5;
    const int lane = tid & 31;
    const int lg   = lane >> 4;           // K-half select per ISA 16-bit A layout
    const int lm   = lane & 15;
    const int row  = (blockIdx.x * 8 + wave) * 16 + lm;   // A row for this lane

    int s = row, py = 0, px = 0;
    if (MODE != 0) {                      // constant-folded division by P / WD
        s = row / P;
        const int p = row - s * P;
        py = p / WD;
        px = p - py * WD;
    }

    v8f acc[NT] = {};

    #pragma unroll
    for (int t = 0; t < TAPS; ++t) {
        // --- stage weights for tap t into LDS --------------------------------
        if (t > 0) __syncthreads();       // all waves done with previous slab
#ifdef HAVE_TDM
        if (tid < 32)
            tdm_load_weights((unsigned)(unsigned long long)(uintptr_t)lw,
                             Wt + t * CIN, CIN, O, KTOT);
#else
        for (int i = tid; i < O * CIN; i += 256)
            lw[i] = Wt[(long)(i / CIN) * KTOT + t * CIN + (i % CIN)];
#endif
        __syncthreads();                  // slab visible to all waves

        // --- per-lane source pixel for tap t ---------------------------------
        const int ky = t / 3, kx = t % 3;
        bool valid = true;
        long arow;
        if (MODE == 0) {
            arow = (long)row * CIN;
        } else if (MODE == 1) {
            const int sy = py - 1 + ky, sx = px - 1 + kx;
            valid = (sy >= 0) && (sy < HD) && (sx >= 0) && (sx < WD);
            arow = ((long)s * P + sy * WD + sx) * CIN;
        } else if (MODE == 2) {
            int sy = py - 1 + ky; if (sy < 0) sy = -sy; if (sy >= HD) sy = 2 * HD - 2 - sy;
            int sx = px - 1 + kx; if (sx < 0) sx = -sx; if (sx >= WD) sx = 2 * WD - 2 - sx;
            arow = ((long)s * P + sy * WD + sx) * CIN;
        } else {                          // transposed conv, stride 2: o = 2*i + k - 1
            const int ty = py + 1 - ky, tx = px + 1 - kx;
            const int iy = ty >> 1,   ix = tx >> 1;
            valid = (ty >= 0) && (tx >= 0) && ((ty & 1) == 0) && ((tx & 1) == 0) &&
                    (iy < HIN) && (ix < WIN);
            arow = ((long)s * (HIN * WIN) + iy * WIN + ix) * CIN;
        }
        const _Float16* ap = A + arow;

        #pragma unroll
        for (int kb = 0; kb < CIN; kb += 32) {
            v16h af = {};
            if (valid) {
                const v8h a_lo = *(const v8h*)(ap + kb + lg * 8);        // K 0..7 / 8..15
                const v8h a_hi = *(const v8h*)(ap + kb + 16 + lg * 8);   // K 16..23 / 24..31
                #pragma unroll
                for (int e = 0; e < 8; ++e) { af[e] = a_lo[e]; af[8 + e] = a_hi[e]; }
            }
            #pragma unroll
            for (int nt = 0; nt < NT; ++nt) {
                const _Float16* wr = lw + (nt * 16 + lm) * CIN + kb;     // LDS fragment
                const v8h b_lo = *(const v8h*)(wr + lg * 8);
                const v8h b_hi = *(const v8h*)(wr + 16 + lg * 8);
                v16h bf;
                #pragma unroll
                for (int e = 0; e < 8; ++e) { bf[e] = b_lo[e]; bf[8 + e] = b_hi[e]; }
                acc[nt] = __builtin_amdgcn_wmma_f32_16x16x32_f16(
                    false, af, false, bf, (short)0, acc[nt], false, false);
            }
        }
        if (valid) __builtin_prefetch(ap + CIN, 0, 1);  // global_prefetch_b8
    }

    // Epilogue. C/D layout: VGPR r -> M = r + 8*lg (lanes 0-15 / 16-31), N = lm.
    const int m0 = (blockIdx.x * 8 + wave) * 16;
    #pragma unroll
    for (int nt = 0; nt < NT; ++nt) {
        const int oc = nt * 16 + lm;
        const float bv = bias[oc];
        #pragma unroll
        for (int r = 0; r < 8; ++r) {
            const long m = m0 + r + 8 * lg;
            float v = acc[nt][r] + bv;
            if (flags & FLAG_RES)   v += resid[m * O + oc];
            if (flags & FLAG_LEAKY) v = (v > 0.f) ? v : LEAKY_SLOPE * v;
            Y[m * O + oc] = v;
        }
    }
}

// Weight repack: f32 (arbitrary OIHW/IOHW strides) -> f16 [O][ (ky*KW+kx)*I + i ]
__global__ void wprep_kernel(const float* __restrict__ W, _Float16* __restrict__ Wt,
                             int O, int I, int KH, int KW, int so, int si, int sh, int sw)
{
    long idx = (long)blockIdx.x * blockDim.x + threadIdx.x;
    long tot = (long)O * I * KH * KW;
    if (idx >= tot) return;
    int per = I * KH * KW;
    int o = (int)(idx / per);
    int r = (int)(idx % per);
    int t = r / I, i = r % I;
    int ky = t / KW, kx = t % KW;
    Wt[idx] = (_Float16)W[(long)o * so + (long)i * si + (long)ky * sh + (long)kx * sw];
}

// Batch-norm statistics (training mode: batch mean + biased var), one block per channel
__global__ void bn_stats_kernel(const float* __restrict__ X, float* __restrict__ mean,
                                float* __restrict__ rstd, int M, int C)
{
    int c = blockIdx.x;
    if (c >= C) return;
    __shared__ float ssum[256], ssq[256];
    float s = 0.f, q = 0.f;
    for (int m = threadIdx.x; m < M; m += blockDim.x) {
        float v = X[(long)m * C + c];
        s += v; q += v * v;
    }
    ssum[threadIdx.x] = s; ssq[threadIdx.x] = q;
    __syncthreads();
    for (int off = 128; off > 0; off >>= 1) {
        if (threadIdx.x < off) {
            ssum[threadIdx.x] += ssum[threadIdx.x + off];
            ssq[threadIdx.x]  += ssq[threadIdx.x + off];
        }
        __syncthreads();
    }
    if (threadIdx.x == 0) {
        float mu = ssum[0] / (float)M;
        float var = ssq[0] / (float)M - mu * mu;
        mean[c] = mu;
        rstd[c] = rsqrtf(var + 1e-5f);
    }
}

// Fused BN (batch stats) + ReLU + f32->f16 staging for the GEMM A operand
__global__ void bn_relu_f16_kernel(const float* __restrict__ X,
                                   const float* __restrict__ mean, const float* __restrict__ rstd,
                                   const float* __restrict__ g, const float* __restrict__ b,
                                   _Float16* __restrict__ Y, long M, int C)
{
    long i = (long)blockIdx.x * blockDim.x + threadIdx.x;
    if (i >= M * C) return;
    int c = (int)(i % C);
    float v = (X[i] - mean[c]) * rstd[c] * g[c] + b[c];
    v = v > 0.f ? v : 0.f;
    Y[i] = (_Float16)v;
}

__global__ void cvt_f16_kernel(const float* __restrict__ X, _Float16* __restrict__ Y, long n)
{
    long i = (long)blockIdx.x * blockDim.x + threadIdx.x;
    if (i < n) Y[i] = (_Float16)X[i];
}

// Input prep: cumulative onehot scatter-add + base offset on channel 0.
// X layout: [s=(b,n,h,w)][d][c], B=2 N=4 C=64 H=W=8 D=144
__global__ void prep_kernel(const float* __restrict__ img, const int* __restrict__ ld,
                            float* __restrict__ X)
{
    long i = (long)blockIdx.x * blockDim.x + threadIdx.x;
    const long tot = 512L * 144 * 64;
    if (i >= tot) return;
    int c = (int)(i & 63);
    int d = (int)((i >> 6) % 144);
    int s = (int)(i / (64 * 144));
    int w = s & 7, h = (s >> 3) & 7, n = (s >> 6) & 3, b = s >> 8;
    float acc = 0.f;
    for (int k = 0; k <= n; ++k) {
        if (ld[b * 4 + k] == d)
            acc += img[((((long)(b * 4 + k)) * 64 + c) * 8 + h) * 8 + w];
    }
    if (c == 0) acc += 1.f - powf(0.9f, (float)(n + 1));
    X[i] = acc;
}

__global__ void maxpool_kernel(const float* __restrict__ X, float* __restrict__ Y,
                               int S, int H, int W)
{
    int Ho = H >> 1, Wo = W >> 1;
    long tot = (long)S * Ho * Wo * 64;
    long i = (long)blockIdx.x * blockDim.x + threadIdx.x;
    if (i >= tot) return;
    int c = (int)(i & 63);
    long q = i >> 6;
    int xo = (int)(q % Wo); q /= Wo;
    int yo = (int)(q % Ho);
    int s  = (int)(q / Ho);
    const float* xp = X + (((long)s * H + 2 * yo) * W + 2 * xo) * 64 + c;
    float a = xp[0], b2 = xp[64], c2 = xp[(long)W * 64], d2 = xp[(long)W * 64 + 64];
    Y[i] = fmaxf(fmaxf(a, b2), fmaxf(c2, d2));
}

// Y = U + nearest_up2(L), optional LeakyReLU
__global__ void upadd_kernel(const float* __restrict__ U, const float* __restrict__ L,
                             float* __restrict__ Y, int S, int H, int W, int leaky)
{
    long tot = (long)S * H * W * 64;
    long i = (long)blockIdx.x * blockDim.x + threadIdx.x;
    if (i >= tot) return;
    int c = (int)(i & 63);
    long q = i >> 6;
    int x = (int)(q % W); q /= W;
    int y = (int)(q % H);
    int s = (int)(q / H);
    int Hh = H >> 1, Wh = W >> 1;
    float v = U[i] + L[(((long)s * Hh + (y >> 1)) * Wh + (x >> 1)) * 64 + c];
    if (leaky) v = v > 0.f ? v : LEAKY_SLOPE * v;
    Y[i] = v;
}

// CBAM channel attention: shared 64->8->64 MLP on mean & max pools, sigmoid scale.
__global__ void __launch_bounds__(64) cbam_channel_kernel(
    const float* __restrict__ X, const float* __restrict__ W1, const float* __restrict__ W2,
    float* __restrict__ OUT, int P)
{
    const int s = blockIdx.x, c = threadIdx.x;
    __shared__ float vmean[64], vmax[64], h[16];
    const float* xp = X + (long)s * P * 64 + c;
    float sm = 0.f, mx = -3.4e38f;
    for (int p = 0; p < P; ++p) { float v = xp[(long)p * 64]; sm += v; if (v > mx) mx = v; }
    vmean[c] = sm / (float)P; vmax[c] = mx;
    __syncthreads();
    if (c < 8)       { float a = 0; for (int j = 0; j < 64; ++j) a += W1[c * 64 + j] * vmean[j];       h[c] = a > 0 ? a : 0; }
    else if (c < 16) { float a = 0; for (int j = 0; j < 64; ++j) a += W1[(c - 8) * 64 + j] * vmax[j];  h[c] = a > 0 ? a : 0; }
    __syncthreads();
    float a = 0.f;
    for (int j = 0; j < 8; ++j) a += W2[c * 8 + j] * (h[j] + h[8 + j]);  // se(mean)+se(max)
    float ca = 1.f / (1.f + expf(-a));
    float* op = OUT + (long)s * P * 64 + c;
    for (int p = 0; p < P; ++p) op[(long)p * 64] = xp[(long)p * 64] * ca;
}

// Spatial attention stats: per pixel [max over C, mean over C]
__global__ void sa_stats_kernel(const float* __restrict__ X, float* __restrict__ SM, long M)
{
    long i = (long)blockIdx.x * blockDim.x + threadIdx.x;
    if (i >= M) return;
    const float* xp = X + i * 64;
    float mx = -3.4e38f, sm = 0.f;
    for (int c = 0; c < 64; ++c) { float v = xp[c]; sm += v; if (v > mx) mx = v; }
    SM[i * 2 + 0] = mx;
    SM[i * 2 + 1] = sm * (1.f / 64.f);
}

// 3x3 zero-pad conv on the 2-channel stat map + sigmoid
__global__ void sa_conv_kernel(const float* __restrict__ SM, const float* __restrict__ w,
                               const float* __restrict__ b, float* __restrict__ SA,
                               int S, int Hd, int Wd)
{
    int P = Hd * Wd;
    long M = (long)S * P;
    long i = (long)blockIdx.x * blockDim.x + threadIdx.x;
    if (i >= M) return;
    int s = (int)(i / P), p = (int)(i % P);
    int py = p / Wd, px = p % Wd;
    float acc = b[0];
    for (int ky = 0; ky < 3; ++ky)
        for (int kx = 0; kx < 3; ++kx) {
            int sy = py - 1 + ky, sx = px - 1 + kx;
            if (sy < 0 || sy >= Hd || sx < 0 || sx >= Wd) continue;
            long q = ((long)s * P + (long)sy * Wd + sx) * 2;
            acc += SM[q + 0] * w[0 * 9 + ky * 3 + kx];
            acc += SM[q + 1] * w[1 * 9 + ky * 3 + kx];
        }
    SA[i] = 1.f / (1.f + expf(-acc));
}

// Y = OUT*SA + RES, then LeakyReLU (post-CBAM activation)
__global__ void cbam_apply_kernel(const float* __restrict__ OUT, const float* __restrict__ SA,
                                  const float* __restrict__ RES, float* __restrict__ Y, long MC)
{
    long i = (long)blockIdx.x * blockDim.x + threadIdx.x;
    if (i >= MC) return;
    float v = OUT[i] * SA[i >> 6] + RES[i];
    Y[i] = v > 0.f ? v : LEAKY_SLOPE * v;
}

// Final 64->1 reflect-pad 3x3 conv
__global__ void c6_kernel(const float* __restrict__ X, const float* __restrict__ w,
                          const float* __restrict__ b, float* __restrict__ out,
                          int S, int Hd, int Wd)
{
    int P = Hd * Wd;
    long M = (long)S * P;
    long i = (long)blockIdx.x * blockDim.x + threadIdx.x;
    if (i >= M) return;
    int s = (int)(i / P), p = (int)(i % P);
    int py = p / Wd, px = p % Wd;
    float acc = b[0];
    for (int ky = 0; ky < 3; ++ky) {
        int sy = py - 1 + ky; if (sy < 0) sy = -sy; if (sy >= Hd) sy = 2 * Hd - 2 - sy;
        for (int kx = 0; kx < 3; ++kx) {
            int sx = px - 1 + kx; if (sx < 0) sx = -sx; if (sx >= Wd) sx = 2 * Wd - 2 - sx;
            const float* xp = X + ((long)s * P + (long)sy * Wd + sx) * 64;
            for (int ic = 0; ic < 64; ++ic)
                acc += xp[ic] * w[ic * 9 + ky * 3 + kx];
        }
    }
    out[i] = acc;
}

__global__ void afmean_kernel(const float* __restrict__ X, float* __restrict__ out, long M)
{
    long i = (long)blockIdx.x * blockDim.x + threadIdx.x;
    if (i >= M) return;
    const float* xp = X + i * 64;
    float sm = 0.f;
    for (int c = 0; c < 64; ++c) sm += xp[c];
    out[i] = sm * (1.f / 64.f);
}

// ---------------------------------------------------------------------------
extern "C" void kernel_launch(void* const* d_in, const int* in_sizes, int n_in,
                              void* d_out, int out_size, void* d_ws, size_t ws_size,
                              hipStream_t stream)
{
    (void)in_sizes; (void)n_in; (void)out_size; (void)ws_size;
    const int  S   = 512;
    const long M12 = 512L * 144, M6 = 512L * 36, M3 = 512L * 9, M24 = 512L * 576;

    auto F = [&](int i) { return (const float*)d_in[i]; };
    // Flattened input order: images, light_dirs, then params as a jax pytree
    // (dict keys sorted alphabetically; lists in order).
    const int IMAGES = 0, LDIRS = 1, C5_B = 2, C5_W = 3, C6_B = 4, C6_W = 5,
              CA_W1 = 6, CA_W2 = 7, HG_BASE = 8,
              SA_B = 344, SA_W = 345, TRANS_B = 346, TRANS_W = 347;
    enum { L_B1 = 0, L_B2, L_B3, L_C1, L_C2, L_C3, L_G1, L_G2, L_G3, L_W1, L_W2, L_W3 };
    auto HG = [&](int lvl, int br, int blk, int leaf) {
        int off = (lvl == 0) ? (br * 4 + blk) : (16 + br * 4 + blk);
        return HG_BASE + off * 12 + leaf;
    };

    // Workspace bump allocator (~350 MB total)
    char* wp = (char*)d_ws;
    auto alloc = [&](size_t bytes) { void* p = wp; wp += (bytes + 255) & ~(size_t)255; return p; };
    float* b12[3]; for (int i = 0; i < 3; ++i) b12[i] = (float*)alloc(M12 * 64 * 4);
    float* b6[2];  for (int i = 0; i < 2; ++i) b6[i]  = (float*)alloc(M6  * 64 * 4);
    float* b3[2];  for (int i = 0; i < 2; ++i) b3[i]  = (float*)alloc(M3  * 64 * 4);
    float* b24[3]; for (int i = 0; i < 3; ++i) b24[i] = (float*)alloc(M24 * 64 * 4);
    float* t1  = (float*)alloc(M12 * 32 * 4);
    float* t2  = (float*)alloc(M12 * 32 * 4);
    _Float16* A16 = (_Float16*)alloc(M24 * 64 * 2);
    _Float16* W16 = (_Float16*)alloc(64 * 576 * 2);
    float* stm = (float*)alloc(64 * 4);
    float* str = (float*)alloc(64 * 4);
    float* SM  = (float*)alloc(M24 * 2 * 4);
    float* SA  = (float*)alloc(M24 * 4);

    auto blocks = [](long n, int b) { return dim3((unsigned)((n + b - 1) / b)); };

    auto wprep = [&](const float* w, int O, int I, int KH, int KW,
                     int so, int si, int sh, int sw) {
        wprep_kernel<<<blocks((long)O * I * KH * KW, 256), 256, 0, stream>>>(
            w, W16, O, I, KH, KW, so, si, sh, sw);
    };

    auto bneck = [&](const float* x, float* y, long M, int Hd, int Wd, int lvl, int br, int blk) {
        dim3 g((unsigned)(M / 128));   // 8 waves x 16 rows per block
        // stage 1: BN+ReLU -> 1x1 conv 64->32
        bn_stats_kernel<<<64, 256, 0, stream>>>(x, stm, str, (int)M, 64);
        bn_relu_f16_kernel<<<blocks(M * 64, 256), 256, 0, stream>>>(
            x, stm, str, F(HG(lvl, br, blk, L_G1)), F(HG(lvl, br, blk, L_B1)), A16, M, 64);
        wprep(F(HG(lvl, br, blk, L_W1)), 32, 64, 1, 1, 64, 1, 1, 1);
        conv_wmma_t<0, 64, 32, 1, 1><<<g, 256, 0, stream>>>(
            A16, W16, F(HG(lvl, br, blk, L_C1)), nullptr, t1, 0);
        // stage 2: BN+ReLU -> 3x3 conv 32->32 (zero pad)
        bn_stats_kernel<<<32, 256, 0, stream>>>(t1, stm, str, (int)M, 32);
        bn_relu_f16_kernel<<<blocks(M * 32, 256), 256, 0, stream>>>(
            t1, stm, str, F(HG(lvl, br, blk, L_G2)), F(HG(lvl, br, blk, L_B2)), A16, M, 32);
        wprep(F(HG(lvl, br, blk, L_W2)), 32, 32, 3, 3, 288, 9, 3, 1);
        if (Hd == 12)
            conv_wmma_t<1, 32, 32, 12, 12><<<g, 256, 0, stream>>>(
                A16, W16, F(HG(lvl, br, blk, L_C2)), nullptr, t2, 0);
        else if (Hd == 6)
            conv_wmma_t<1, 32, 32, 6, 6><<<g, 256, 0, stream>>>(
                A16, W16, F(HG(lvl, br, blk, L_C2)), nullptr, t2, 0);
        else
            conv_wmma_t<1, 32, 32, 3, 3><<<g, 256, 0, stream>>>(
                A16, W16, F(HG(lvl, br, blk, L_C2)), nullptr, t2, 0);
        // stage 3: BN+ReLU -> 1x1 conv 32->64 + residual
        bn_stats_kernel<<<32, 256, 0, stream>>>(t2, stm, str, (int)M, 32);
        bn_relu_f16_kernel<<<blocks(M * 32, 256), 256, 0, stream>>>(
            t2, stm, str, F(HG(lvl, br, blk, L_G3)), F(HG(lvl, br, blk, L_B3)), A16, M, 32);
        wprep(F(HG(lvl, br, blk, L_W3)), 64, 32, 1, 1, 32, 1, 1, 1);
        conv_wmma_t<0, 32, 64, 1, 1><<<g, 256, 0, stream>>>(
            A16, W16, F(HG(lvl, br, blk, L_C3)), x, y, FLAG_RES);
    };

    auto residual = [&](float** cur, float** alt, long M, int Hd, int Wd, int lvl, int br) {
        for (int blk = 0; blk < 4; ++blk) {
            bneck(*cur, *alt, M, Hd, Wd, lvl, br, blk);
            float* tp = *cur; *cur = *alt; *alt = tp;
        }
    };

    // 0) Input prep -> x12 in b12[0]
    prep_kernel<<<blocks(M12 * 64, 256), 256, 0, stream>>>(
        F(IMAGES), (const int*)d_in[LDIRS], b12[0]);

    // 1) Hourglass depth 2
    float *cur12 = b12[0], *alt12 = b12[1];
    float *cur6 = b6[0], *alt6 = b6[1];
    float *cur3 = b3[0], *alt3 = b3[1];

    maxpool_kernel<<<blocks(M6 * 64, 256), 256, 0, stream>>>(cur12, cur6, S, 12, 12);
    residual(&cur12, &alt12, M12, 12, 12, 1, 0);                 // up1 @12
    residual(&cur6,  &alt6,  M6,  6,  6,  1, 1);                 // low1 @6
    maxpool_kernel<<<blocks(M3 * 64, 256), 256, 0, stream>>>(cur6, cur3, S, 6, 6);
    residual(&cur6,  &alt6,  M6,  6,  6,  0, 0);                 // inner up1 @6
    residual(&cur3,  &alt3,  M3,  3,  3,  0, 1);                 // inner low1 @3
    residual(&cur3,  &alt3,  M3,  3,  3,  0, 3);                 // innermost @3
    residual(&cur3,  &alt3,  M3,  3,  3,  0, 2);                 // inner low3 @3
    upadd_kernel<<<blocks(M6 * 64, 256), 256, 0, stream>>>(cur6, cur3, alt6, S, 6, 6, 0);
    { float* tp = cur6; cur6 = alt6; alt6 = tp; }                // low2 @6
    residual(&cur6,  &alt6,  M6,  6,  6,  1, 2);                 // low3 @6
    upadd_kernel<<<blocks(M12 * 64, 256), 256, 0, stream>>>(cur12, cur6, b12[2], S, 12, 12, 1);
    float* hx = b12[2];                                          // leaky(hourglass out)

    // 2) Transposed conv 64->64, 12x12 -> 24x24 (IOHW weights), fused leaky
    cvt_f16_kernel<<<blocks(M12 * 64, 256), 256, 0, stream>>>(hx, A16, M12 * 64);
    wprep(F(TRANS_W), 64, 64, 3, 3, 9, 576, 3, 1);
    conv_wmma_t<3, 64, 64, 24, 24><<<dim3((unsigned)(M24 / 128)), 256, 0, stream>>>(
        A16, W16, F(TRANS_B), nullptr, b24[0], FLAG_LEAKY);

    // 3) c5 reflect-pad 3x3 conv 64->64, fused leaky
    cvt_f16_kernel<<<blocks(M24 * 64, 256), 256, 0, stream>>>(b24[0], A16, M24 * 64);
    wprep(F(C5_W), 64, 64, 3, 3, 576, 9, 3, 1);
    conv_wmma_t<2, 64, 64, 24, 24><<<dim3((unsigned)(M24 / 128)), 256, 0, stream>>>(
        A16, W16, F(C5_B), nullptr, b24[1], FLAG_LEAKY);

    // 4) CBAM + post-leaky
    cbam_channel_kernel<<<512, 64, 0, stream>>>(b24[1], F(CA_W1), F(CA_W2), b24[2], 576);
    sa_stats_kernel<<<blocks(M24, 256), 256, 0, stream>>>(b24[2], SM, M24);
    sa_conv_kernel<<<blocks(M24, 256), 256, 0, stream>>>(SM, F(SA_W), F(SA_B), SA, S, 24, 24);
    cbam_apply_kernel<<<blocks(M24 * 64, 256), 256, 0, stream>>>(b24[2], SA, b24[1], b24[0], M24 * 64);

    // 5) Outputs: c6 conv map then channel-mean feature map
    float* out = (float*)d_out;
    c6_kernel<<<blocks(M24, 128), 128, 0, stream>>>(b24[0], F(C6_W), F(C6_B), out, S, 24, 24);
    afmean_kernel<<<blocks(M24, 256), 256, 0, stream>>>(b24[0], out + M24, M24);
}
